// RAUMoE_9577777070296
// MI455X (gfx1250) — matrix operand order repaired and checked by reading, not compiled, not run on hardware
//
#include <hip/hip_runtime.h>

#define NTOK   32768
#define DIM    512
#define NEXP   8
#define NLAY   3            // ReLU D->D layers; + 1 output layer
#define TM     64
#define HSTRIDE 520         // 512 + 8 bf16 pad (16B-aligned rows, conflict-breaking)
#define WSTRIDE 40          // 32 + 8 bf16 pad (80B row stride, 16B aligned)
#define TILES  (NTOK / TM)  // 512

typedef __attribute__((ext_vector_type(16))) __bf16 v16bf;
typedef __attribute__((ext_vector_type(8)))  float  v8f;

union Frag16 { uint4 u4[2]; v16bf v; };

__device__ __forceinline__ unsigned short f32_to_bf16_rne(float f) {
  union { float f; unsigned int u; } c; c.f = f;
  unsigned int u = c.u;
  u += 0x7FFFu + ((u >> 16) & 1u);   // round-to-nearest-even
  return (unsigned short)(u >> 16);
}

// ---------------------------------------------------------------------------
// Weight transpose + fp32 -> bf16 convert:  dst[mat][n][k] = bf16(src[mat][k][n])
// ---------------------------------------------------------------------------
__global__ __launch_bounds__(256)
void wconv_kernel(const float* __restrict__ src, unsigned short* __restrict__ dst)
{
  __shared__ float tile[64][65];
  const int mat = blockIdx.x >> 6;
  const int t   = blockIdx.x & 63;
  const int tx  = (t & 7) * 64;    // src col (n)
  const int ty  = (t >> 3) * 64;   // src row (k)
  const float* S = src + (size_t)mat * DIM * DIM;
  unsigned short* Dst = dst + (size_t)mat * DIM * DIM;

  for (int i = threadIdx.x; i < 64 * 16; i += 256) {
    const int r = i >> 4, c = (i & 15) * 4;
    const float4 v = *(const float4*)(S + (size_t)(ty + r) * DIM + tx + c);
    tile[r][c] = v.x; tile[r][c + 1] = v.y; tile[r][c + 2] = v.z; tile[r][c + 3] = v.w;
  }
  __syncthreads();
  for (int i = threadIdx.x; i < 64 * 16; i += 256) {
    const int r = i >> 4, c = (i & 15) * 4;  // r: n-offset, c: k-offset
    ushort4 o;
    o.x = f32_to_bf16_rne(tile[c][r]);
    o.y = f32_to_bf16_rne(tile[c + 1][r]);
    o.z = f32_to_bf16_rne(tile[c + 2][r]);
    o.w = f32_to_bf16_rne(tile[c + 3][r]);
    *(ushort4*)(Dst + (size_t)(tx + r) * DIM + ty + c) = o;
  }
}

// ---------------------------------------------------------------------------
// Routing: top-2 over 8 gates from the last 3 features, softmax, atomic
// compaction into per-expert token lists.
// ---------------------------------------------------------------------------
__global__ void moe_route_kernel(const float* __restrict__ x,
                                 const float* __restrict__ Wg,  // [3][E]
                                 const float* __restrict__ bg,  // [E]
                                 int* __restrict__ counts,
                                 int* __restrict__ tokIdx,
                                 float* __restrict__ tokW)
{
  const int n = blockIdx.x * blockDim.x + threadIdx.x;
  if (n >= NTOK) return;
  const float x0 = x[(size_t)n * DIM + DIM - 3];
  const float x1 = x[(size_t)n * DIM + DIM - 2];
  const float x2 = x[(size_t)n * DIM + DIM - 1];
  float g[NEXP];
#pragma unroll
  for (int e = 0; e < NEXP; ++e)
    g[e] = x0 * Wg[e] + x1 * Wg[NEXP + e] + x2 * Wg[2 * NEXP + e] + bg[e];

  int i0 = 0; float v0 = g[0];
#pragma unroll
  for (int e = 1; e < NEXP; ++e) if (g[e] > v0) { v0 = g[e]; i0 = e; }
  int i1 = -1; float v1 = -1e30f;
#pragma unroll
  for (int e = 0; e < NEXP; ++e) if (e != i0 && g[e] > v1) { v1 = g[e]; i1 = e; }

  const float e1 = __expf(v1 - v0);
  const float w0 = 1.0f / (1.0f + e1);
  const float w1 = e1 * w0;

  int s0 = atomicAdd(&counts[i0], 1);
  tokIdx[i0 * NTOK + s0] = n;  tokW[i0 * NTOK + s0] = w0;
  int s1 = atomicAdd(&counts[i1], 1);
  tokIdx[i1 * NTOK + s1] = n;  tokW[i1 * NTOK + s1] = w1;
}

// ---------------------------------------------------------------------------
// Fused expert MLP. 8 waves as 2(M) x 4(N); wave owns 32 rows x 128 cols.
// Weight K-slices: global -> registers (8 b128 in flight) -> LDS double
// buffer; one barrier per K-step; WMMAs overlap next slice's global loads.
// ---------------------------------------------------------------------------
__global__ __launch_bounds__(256)
void moe_expert_kernel(const float* __restrict__ x,
                       const float* __restrict__ bh,   // [E][NLAY][D]
                       const float* __restrict__ bo,   // [E][D]
                       const unsigned short* __restrict__ WhT,  // [E][NLAY][n][k] bf16
                       const unsigned short* __restrict__ WoT,  // [E][n][k] bf16
                       const int* __restrict__ counts,
                       const int* __restrict__ tokIdx,
                       const float* __restrict__ tokW,
                       float* __restrict__ out)
{
  __shared__ unsigned short hbuf[2][TM][HSTRIDE];      // activations, double buffered
  __shared__ unsigned short wslice[2][DIM][WSTRIDE];   // K=32 weight slices, [n][k]
  __shared__ int   stok[TM];
  __shared__ float sw[TM];

  const int e    = blockIdx.x / TILES;
  const int tile = blockIdx.x % TILES;
  const int base = tile * TM;
  const int cnt  = counts[e];
  if (base >= cnt) return;

  const int tid  = threadIdx.x;
  const int lane = tid & 31;
  const int wave = tid >> 5;
  const int r0   = (wave & 1) * 32;    // M-group
  const int c0   = (wave >> 1) * 128;  // N-group

  if (tid < TM) {
    const int i  = base + tid;
    const int ok = (i < cnt);
    stok[tid] = ok ? tokIdx[e * NTOK + i] : 0;
    sw[tid]   = ok ? tokW[e * NTOK + i] : 0.0f;
  }
  __syncthreads();

  // Gather token rows -> bf16 hbuf[0]; batch 8 loads in flight at a time.
  {
    const int rbase = tid >> 7;      // (tid + 256*m) >> 7 == 2m + rbase
    const int j     = tid & 127;
#pragma unroll
    for (int b = 0; b < 4; ++b) {
      float4 v[8];
#pragma unroll
      for (int i = 0; i < 8; ++i) {
        const int r = rbase + 2 * (b * 8 + i);
        v[i] = *(const float4*)(x + (size_t)stok[r] * DIM + j * 4);
      }
#pragma unroll
      for (int i = 0; i < 8; ++i) {
        const int r = rbase + 2 * (b * 8 + i);
        ushort4 o;
        o.x = f32_to_bf16_rne(v[i].x); o.y = f32_to_bf16_rne(v[i].y);
        o.z = f32_to_bf16_rne(v[i].z); o.w = f32_to_bf16_rne(v[i].w);
        *(ushort4*)&hbuf[0][r][j << 2] = o;
      }
    }
  }

  int cur = 0;
  const int aklo = (lane < 16) ? 0 : 8;    // A frag: elems 0-7 = K aklo.., 8-15 = K aklo+16..
  const int bklo = (lane < 16) ? 0 : 8;    // B frag: elems 0-7 = K bklo.., 8-15 = K bklo+16..
  const int l16  = lane & 15;
  const int rhi  = (lane < 16) ? 0 : 8;    // C frag row offset
  const int wn   = tid >> 2;               // this thread's slice row
  const int wpart = tid & 3;               // 16B chunk within 64B k-run

  for (int layer = 0; layer <= NLAY; ++layer) {
    const bool last = (layer == NLAY);
    const unsigned short* W = last ? (WoT + (size_t)e * DIM * DIM)
                                   : (WhT + (size_t)(e * NLAY + layer) * DIM * DIM);
    const float* bias = last ? (bo + (size_t)e * DIM)
                             : (bh + (size_t)(e * NLAY + layer) * DIM);

    const v8f vzero = {0.f, 0.f, 0.f, 0.f, 0.f, 0.f, 0.f, 0.f};
    v8f acc[2][8];
#pragma unroll
    for (int i = 0; i < 2; ++i)
#pragma unroll
      for (int j = 0; j < 8; ++j) acc[i][j] = vzero;

    uint4 wreg[8];
    // prologue: slice 0 -> regs -> wslice[0]
#pragma unroll
    for (int i = 0; i < 8; ++i)
      wreg[i] = *(const uint4*)(W + (size_t)(wn + i * 64) * DIM + wpart * 8);
#pragma unroll
    for (int i = 0; i < 8; ++i)
      *(uint4*)&wslice[0][wn + i * 64][wpart << 3] = wreg[i];
    __syncthreads();

    for (int ks = 0; ks < DIM / 32; ++ks) {
      const int buf = ks & 1;
      // issue next slice's global loads before compute (hide behind WMMAs)
      if (ks < DIM / 32 - 1) {
#pragma unroll
        for (int i = 0; i < 8; ++i)
          wreg[i] = *(const uint4*)(W + (size_t)(wn + i * 64) * DIM
                                      + (ks + 1) * 32 + wpart * 8);
      }

      // A fragments from activation LDS
      Frag16 a0, a1;
      const unsigned short* ap0 = &hbuf[cur][r0 + l16][ks * 32 + aklo];
      const unsigned short* ap1 = &hbuf[cur][r0 + 16 + l16][ks * 32 + aklo];
      a0.u4[0] = *(const uint4*)ap0;  a0.u4[1] = *(const uint4*)(ap0 + 16);
      a1.u4[0] = *(const uint4*)ap1;  a1.u4[1] = *(const uint4*)(ap1 + 16);

      // B fragments pipelined one ahead of their WMMAs
      Frag16 bcur, bnxt;
      {
        const unsigned short* bp = &wslice[buf][c0 + l16][bklo];
        bcur.u4[0] = *(const uint4*)bp;  bcur.u4[1] = *(const uint4*)(bp + 16);
      }
#pragma unroll
      for (int nf = 0; nf < 8; ++nf) {
        if (nf < 7) {
          const unsigned short* bp = &wslice[buf][c0 + (nf + 1) * 16 + l16][bklo];
          bnxt.u4[0] = *(const uint4*)bp;  bnxt.u4[1] = *(const uint4*)(bp + 16);
        }
        acc[0][nf] = __builtin_amdgcn_wmma_f32_16x16x32_bf16(
            false, a0.v, false, bcur.v, (short)0, acc[0][nf], false, false);
        acc[1][nf] = __builtin_amdgcn_wmma_f32_16x16x32_bf16(
            false, a1.v, false, bcur.v, (short)0, acc[1][nf], false, false);
        bcur = bnxt;
      }

      // park next slice into the other LDS buffer
      if (ks < DIM / 32 - 1) {
#pragma unroll
        for (int i = 0; i < 8; ++i)
          *(uint4*)&wslice[buf ^ 1][wn + i * 64][wpart << 3] = wreg[i];
      }
      __syncthreads();
    }

    if (!last) {
#pragma unroll
      for (int i = 0; i < 2; ++i)
#pragma unroll
        for (int nf = 0; nf < 8; ++nf) {
          const int col = c0 + nf * 16 + l16;
          const float bv = bias[col];
#pragma unroll
          for (int v = 0; v < 8; ++v) {
            float f = acc[i][nf][v] + bv;
            f = fmaxf(f, 0.0f);
            hbuf[cur ^ 1][r0 + i * 16 + rhi + v][col] = f32_to_bf16_rne(f);
          }
        }
      cur ^= 1;
    } else {
#pragma unroll
      for (int i = 0; i < 2; ++i)
#pragma unroll
        for (int nf = 0; nf < 8; ++nf) {
          const int col = c0 + nf * 16 + l16;
          const float bv = bias[col];
#pragma unroll
          for (int v = 0; v < 8; ++v) {
            const int row = r0 + i * 16 + rhi + v;
            if (base + row < cnt) {
              const float y = acc[i][nf][v] + bv;
              atomicAdd(&out[(size_t)stok[row] * DIM + col], sw[row] * y);
            }
          }
        }
    }
  }
}

// ---------------------------------------------------------------------------
extern "C" void kernel_launch(void* const* d_in, const int* in_sizes, int n_in,
                              void* d_out, int out_size, void* d_ws, size_t ws_size,
                              hipStream_t stream)
{
  const float* x  = (const float*)d_in[0];
  const float* Wg = (const float*)d_in[1];
  const float* bg = (const float*)d_in[2];
  const float* Wh = (const float*)d_in[3];
  const float* bh = (const float*)d_in[4];
  const float* Wo = (const float*)d_in[5];
  const float* bo = (const float*)d_in[6];
  float* out = (float*)d_out;

  char* ws = (char*)d_ws;
  int*   counts = (int*)ws;                                     // 256 B
  int*   tokIdx = (int*)(ws + 256);                             // E*N*4 = 1 MB
  float* tokW   = (float*)(ws + 256 + (size_t)NEXP * NTOK * 4); // 1 MB
  unsigned short* WhT = (unsigned short*)(ws + 256 + 2 * (size_t)NEXP * NTOK * 4);
  unsigned short* WoT = WhT + (size_t)NEXP * NLAY * DIM * DIM;  // +12.6 MB, then 4.2 MB

  hipMemsetAsync(out, 0, (size_t)out_size * sizeof(float), stream);
  hipMemsetAsync(counts, 0, 256, stream);
  wconv_kernel<<<NEXP * NLAY * 64, 256, 0, stream>>>(Wh, WhT);
  wconv_kernel<<<NEXP * 64, 256, 0, stream>>>(Wo, WoT);
  moe_route_kernel<<<NTOK / 256, 256, 0, stream>>>(x, Wg, bg, counts, tokIdx, tokW);
  moe_expert_kernel<<<NEXP * TILES, 256, 0, stream>>>(x, bh, bo, WhT, WoT,
                                                      counts, tokIdx, tokW, out);
  (void)in_sizes; (void)n_in; (void)ws_size;
}
